// RGINConv_54400055771236
// MI455X (gfx1250) — matrix operand order repaired and checked by reading, not compiled, not run on hardware
//
#include <hip/hip_runtime.h>
#include <hip/hip_bf16.h>

// ---------------------------------------------------------------------------
// R-GIN: out = feat + segment_sum_dst( feat[src] @ W[etype] )
// Strategy: bucket edges by relation, then grouped GEMM with bf16 WMMA
// (V_WMMA_F32_16X16X32_BF16), f32 accumulate, f32 atomic scatter-add.
// ---------------------------------------------------------------------------

typedef __attribute__((ext_vector_type(16))) __bf16 bf16x16;
typedef __attribute__((ext_vector_type(8)))  float  f32x8;

#define DFEAT 128

// meta layout (ints): [0..7]=counts, [8..15]=cursor, [16..23]=bucketOffset,
// [24..24+R]=groupStart (groupStart[R]=totalGroups)

__global__ void rgin_init_meta(int* meta) {
    if (threadIdx.x < 40) meta[threadIdx.x] = 0;
}

__global__ void rgin_count(const int* __restrict__ et, int E, int* __restrict__ meta) {
    int i = blockIdx.x * blockDim.x + threadIdx.x;
    if (i < E) atomicAdd(&meta[et[i]], 1);
}

__global__ void rgin_scan(int* meta, int R) {
    // single thread: tiny R (=8)
    int off = 0, goff = 0;
    for (int r = 0; r < R; ++r) {
        int c = meta[r];
        meta[16 + r] = off;   // bucketOffset
        meta[8 + r]  = off;   // scatter cursor
        meta[24 + r] = goff;  // groupStart
        off  += c;
        goff += (c + 15) >> 4;
    }
    meta[24 + R] = goff;      // total groups
}

__global__ void rgin_scatter(const int* __restrict__ et, int E,
                             int* __restrict__ meta, int* __restrict__ bucket) {
    int i = blockIdx.x * blockDim.x + threadIdx.x;
    if (i < E) {
        int p = atomicAdd(&meta[8 + et[i]], 1);
        bucket[p] = i;
    }
}

// out = feat  ((1+eps)*h with eps==0)
__global__ void rgin_init_out(const float4* __restrict__ feat, float4* __restrict__ out, int n4) {
    int i = blockIdx.x * blockDim.x + threadIdx.x;
    if (i < n4) out[i] = feat[i];
}

// Wt[r][n][k] = (bf16) W[r][k][n]   (B-matrix becomes contiguous-in-K per lane)
__global__ void rgin_convW(const float* __restrict__ W, __bf16* __restrict__ Wt, int total) {
    int i = blockIdx.x * blockDim.x + threadIdx.x;
    if (i < total) {
        int r   = i >> 14;          // / (128*128)
        int rem = i & 16383;
        int k   = rem >> 7;
        int n   = rem & 127;
        Wt[(((size_t)r << 7 | n) << 7) | k] = (__bf16)W[i];
    }
}

// One wave (32 lanes) per group of up to 16 edges sharing one relation.
// D(16x128) = A(16x128 bf16 gathered feats) @ W_r(128x128 bf16), f32 acc.
__global__ void __launch_bounds__(128)
rgin_gemm(const float* __restrict__ feat, const __bf16* __restrict__ Wt,
          const int* __restrict__ src, const int* __restrict__ dst,
          const int* __restrict__ meta, const int* __restrict__ bucket,
          float* __restrict__ out, int R) {
    const int lane = threadIdx.x & 31;
    const int g = blockIdx.x * (blockDim.x >> 5) + (threadIdx.x >> 5);
    const int total = meta[24 + R];
    if (g >= total) return;                       // wave-uniform: EXEC all-1 below

    // map group -> relation (groupStart is nondecreasing, R==8 so linear scan)
    int r = 0;
    while (r + 1 < R && g >= meta[24 + r + 1]) ++r;
    const int localG  = g - meta[24 + r];
    const int cnt     = min(16, meta[r] - localG * 16);
    const int grpBase = meta[16 + r] + localG * 16;

    const int half = lane >> 4;      // K-half selector per WMMA 16-bit layout
    const int nlo  = lane & 15;      // A row / B column / D column owned by lane

    // A: this lane supplies row m = nlo (both halves supply disjoint K ranges)
    const int eA   = bucket[grpBase + ((nlo < cnt) ? nlo : 0)];
    const float* arow = feat + (size_t)src[eA] * DFEAT;

    const __bf16* wrel = Wt + ((size_t)r << 14);  // r*128*128

    f32x8 acc[8] = {};                            // 8 N-tiles of 16x16 f32

#pragma unroll
    for (int kc = 0; kc < 4; ++kc) {              // K chunks of 32
        // ---- A fragment: elems i -> K = kc*32 + 16*half + i
        const int kA = kc * 32 + half * 16;
        f32x8 f0 = *(const f32x8*)(arow + kA);
        f32x8 f1 = *(const f32x8*)(arow + kA + 8);
        bf16x16 a;
#pragma unroll
        for (int i = 0; i < 8; ++i) { a[i] = (__bf16)f0[i]; a[8 + i] = (__bf16)f1[i]; }

        // ---- B fragments: lane = column n, elems i -> K = kc*32 + 16*half + i
        const int kB = kc * 32 + half * 16;
#pragma unroll
        for (int nt = 0; nt < 8; ++nt) {
            const __bf16* brow = wrel + (((size_t)(nt * 16 + nlo)) << 7) + kB;
            bf16x16 b = *(const bf16x16*)brow;    // 32B contiguous
            acc[nt] = __builtin_amdgcn_wmma_f32_16x16x32_bf16(
                false, a, false, b, (short)0, acc[nt], false, false);
        }
    }

    // D layout: element v -> row M = v + 8*half, col N = nt*16 + nlo
#pragma unroll
    for (int v = 0; v < 8; ++v) {
        const int m = v + half * 8;
        if (m < cnt) {
            const int e = bucket[grpBase + m];
            float* orow = out + (size_t)dst[e] * DFEAT + nlo;
#pragma unroll
            for (int nt = 0; nt < 8; ++nt)
                unsafeAtomicAdd(orow + nt * 16, acc[nt][v]);
        }
    }
}

extern "C" void kernel_launch(void* const* d_in, const int* in_sizes, int n_in,
                              void* d_out, int out_size, void* d_ws, size_t ws_size,
                              hipStream_t stream) {
    const float* feat   = (const float*)d_in[0];
    const float* W      = (const float*)d_in[1];
    const int*   etypes = (const int*)d_in[2];
    const int*   src    = (const int*)d_in[3];
    const int*   dst    = (const int*)d_in[4];
    float*       out    = (float*)d_out;

    const int E = in_sizes[2];
    const int R = in_sizes[1] / (DFEAT * DFEAT);
    const int nOut4 = out_size / 4;

    // workspace layout
    char* ws = (char*)d_ws;
    int*  meta   = (int*)ws;                                  // 40 ints
    int*  bucket = (int*)(ws + 256);                          // E ints
    size_t wtOff = (256 + (size_t)E * sizeof(int) + 255) & ~(size_t)255;
    __bf16* Wt   = (__bf16*)(ws + wtOff);                     // R*128*128 bf16
    // total ws needed: ~2.6 MB + 256 KB

    // 1) bucketing pipeline
    rgin_init_meta<<<1, 64, 0, stream>>>(meta);
    rgin_count<<<(E + 255) / 256, 256, 0, stream>>>(etypes, E, meta);
    rgin_scan<<<1, 1, 0, stream>>>(meta, R);
    rgin_scatter<<<(E + 255) / 256, 256, 0, stream>>>(etypes, E, meta, bucket);

    // 2) out = feat  (eps = 0)
    rgin_init_out<<<(nOut4 + 255) / 256, 256, 0, stream>>>(
        (const float4*)feat, (float4*)out, nOut4);

    // 3) W -> bf16 transposed
    const int wTot = R * DFEAT * DFEAT;
    rgin_convW<<<(wTot + 255) / 256, 256, 0, stream>>>(W, Wt, wTot);

    // 4) grouped WMMA GEMM + scatter-add
    const int maxGroups = (E + 15) / 16 + R;   // upper bound on sum of ceil(cnt_r/16)
    const int wavesPerBlk = 4;                 // blockDim 128 = 4 wave32
    const int blocks = (maxGroups + wavesPerBlk - 1) / wavesPerBlk;
    rgin_gemm<<<blocks, 128, 0, stream>>>(feat, Wt, src, dst, meta, bucket, out, R);
}
